// Attention_21672404976189
// MI455X (gfx1250) — compile-verified, matrix operand order checked by reading, not run on hardware
//
#include <hip/hip_runtime.h>
#include <hip/hip_bf16.h>

// ---------------------------------------------------------------------------
// Problem constants (B=2, S=2048, H=2048, 16 heads, head_dim=128)
// ---------------------------------------------------------------------------
#define BATCH 2
#define SEQ   2048
#define HID   2048
#define NH    16
#define HD    128
#define ROWS  (BATCH * SEQ)          // 4096
#define NEGM  (-1e4f)

typedef __attribute__((ext_vector_type(16))) __bf16 bf16x16;
typedef __attribute__((ext_vector_type(8)))  float  f32x8;

union FragBF {
    bf16x16 v;
    uint4   q[2];
};
union Half8 {
    uint4  q;
    __bf16 h[8];
};

// ---------------------------------------------------------------------------
// Fragment loaders.
// A-matrix (16x32 bf16): lane L holds row M = L%16; lanes<16 hold K chunks
// {0..7, 16..23}, lanes>=16 hold {8..15, 24..31}. B-matrix mirrors with
// lane -> N (column) index, column data contiguous along K.
// ---------------------------------------------------------------------------
__device__ __forceinline__ bf16x16 ldsFrag(const __bf16* tile, int ld,
                                           int rowBase, int col0, int lane) {
    const int r  = rowBase + (lane & 15);
    const int kb = (lane & 16) ? 8 : 0;
    const __bf16* p = tile + r * ld + col0 + kb;
    FragBF f;
    f.q[0] = *(const uint4*)(p);
    f.q[1] = *(const uint4*)(p + 16);
    return f.v;
}

__device__ __forceinline__ bf16x16 gFrag(const __bf16* base, size_t ld,
                                         int row0, int col0, int lane) {
    const int r  = row0 + (lane & 15);
    const int kb = (lane & 16) ? 8 : 0;
    const __bf16* p = base + (size_t)r * ld + col0 + kb;
    FragBF f;
    f.q[0] = *(const uint4*)(p);
    f.q[1] = *(const uint4*)(p + 16);
    return f.v;
}

__device__ __forceinline__ f32x8 wmma_bf16(bf16x16 a, bf16x16 b, f32x8 c) {
    return __builtin_amdgcn_wmma_f32_16x16x32_bf16(
        false, a, false, b, (short)0, c, false, false);
}

__device__ __forceinline__ float rowMax16(float x) {
#pragma unroll
    for (int m = 1; m < 16; m <<= 1) x = fmaxf(x, __shfl_xor(x, m, 16));
    return x;
}
__device__ __forceinline__ float rowSum16(float x) {
#pragma unroll
    for (int m = 1; m < 16; m <<= 1) x += __shfl_xor(x, m, 16);
    return x;
}

// ---------------------------------------------------------------------------
// Async (CDNA5) tile staging: copy a 128x32 bf16 tile global -> LDS through
// GLOBAL_LOAD_ASYNC_TO_LDS_B128 (no VGPR data, tracked by ASYNCcnt).
// One instruction moves 32 lanes x 16B = 512B; tile = 16 instrs block-wide
// (2 per wave). Low 32 bits of a generic shared pointer are the wave-relative
// LDS offset (flat LDS aperture), which is what the instruction's VDST wants.
// ---------------------------------------------------------------------------
__device__ __forceinline__ void async_tile(const __bf16* __restrict__ g, int ld,
                                           const __bf16* l, int t) {
#pragma unroll
    for (int i = 0; i < 2; ++i) {
        const int c   = i * 256 + t;       // 0..511 sixteen-byte chunks
        const int row = c >> 2;            // 0..127
        const int qn  = c & 3;             // 4 chunks per 32-elem row
        const uint64_t ga = (uint64_t)(uintptr_t)(g + (size_t)row * ld + qn * 8);
        const uint32_t la = (uint32_t)(uintptr_t)(l + row * 32 + qn * 8);
        asm volatile("global_load_async_to_lds_b128 %0, %1, off"
                     :: "v"(la), "v"(ga) : "memory");
    }
}

// ---------------------------------------------------------------------------
// One-time f32 -> bf16 conversion (X and the four weight matrices) so the
// GEMM hot loops stream half the bytes and stage without VALU converts.
// ---------------------------------------------------------------------------
__global__ __launch_bounds__(256) void cvt_kernel(const float* __restrict__ in,
                                                  __bf16* __restrict__ out) {
    const int i = blockIdx.x * blockDim.x + threadIdx.x;   // one float4 each
    const float4 f = ((const float4*)in)[i];
    union { __bf16 h[4]; uint2 u; } p;
    p.h[0] = (__bf16)f.x;
    p.h[1] = (__bf16)f.y;
    p.h[2] = (__bf16)f.z;
    p.h[3] = (__bf16)f.w;
    ((uint2*)out)[i] = p.u;
}

// ---------------------------------------------------------------------------
// GEMM: C[M,N] = A[M,K] @ B[N,K]^T (F.linear), A and B bf16, C bf16 or f32.
// Block 128x128, 256 threads = 8 waves; wave computes 64x32 (4x2 frags).
// Double-buffered LDS staged by async-to-LDS; ASYNCcnt-pipelined.
// ---------------------------------------------------------------------------
template <bool OUT_F32>
__global__ __launch_bounds__(256) void gemm_kernel(const __bf16* __restrict__ A,
                                                   const __bf16* __restrict__ Bw,
                                                   void* __restrict__ C,
                                                   int M, int N, int K) {
    __shared__ __bf16 As[2][128 * 32];
    __shared__ __bf16 Bs[2][128 * 32];

    const int t    = threadIdx.x;
    const int lane = t & 31;
    const int w    = t >> 5;
    const int wm   = w & 1;        // 0..1  -> 64-row slice
    const int wn   = w >> 1;       // 0..3  -> 32-col slice
    const int bm   = blockIdx.y * 128;
    const int bn   = blockIdx.x * 128;

    const __bf16* Arow = A  + (size_t)bm * K;
    const __bf16* Brow = Bw + (size_t)bn * K;

    f32x8 acc[4][2] = {};
    const int KT = K / 32;

    // Prologue stage of buffer 0
    async_tile(Arow, K, As[0], t);
    async_tile(Brow, K, Bs[0], t);
    asm volatile("s_wait_asynccnt 0x0" ::: "memory");
    __syncthreads();

    for (int kt = 0; kt < KT; ++kt) {
        const int  buf  = kt & 1;
        const bool more = (kt + 1) < KT;
        if (more) {
            async_tile(Arow + (kt + 1) * 32, K, As[buf ^ 1], t);
            async_tile(Brow + (kt + 1) * 32, K, Bs[buf ^ 1], t);
        }

        bf16x16 af[4], bf[2];
#pragma unroll
        for (int i = 0; i < 4; ++i)
            af[i] = ldsFrag(As[buf], 32, wm * 64 + i * 16, 0, lane);
#pragma unroll
        for (int j = 0; j < 2; ++j)
            bf[j] = ldsFrag(Bs[buf], 32, wn * 32 + j * 16, 0, lane);
#pragma unroll
        for (int i = 0; i < 4; ++i)
#pragma unroll
            for (int j = 0; j < 2; ++j)
                acc[i][j] = wmma_bf16(af[i], bf[j], acc[i][j]);

        if (more) {
            asm volatile("s_wait_asynccnt 0x0" ::: "memory");
            __syncthreads();
        }
    }

    // Epilogue: C layout -> VGPR r holds rows {r, r+8}; lane%16 = column.
    const int moff = (lane & 16) ? 8 : 0;
    const int cl   = lane & 15;
#pragma unroll
    for (int i = 0; i < 4; ++i)
#pragma unroll
        for (int j = 0; j < 2; ++j)
#pragma unroll
            for (int r = 0; r < 8; ++r) {
                const int row = bm + wm * 64 + i * 16 + r + moff;
                const int col = bn + wn * 32 + j * 16 + cl;
                const float v = acc[i][j][r];
                if (OUT_F32)
                    ((float*)C)[(size_t)row * N + col] = v;
                else
                    ((__bf16*)C)[(size_t)row * N + col] = (__bf16)v;
            }
}

// ---------------------------------------------------------------------------
// RoPE, in-place on bf16 [ROWS, HID] viewed as [B,S,NH,HD].
// Each thread owns the (d, d+64) pair so in-place is race-free.
// ---------------------------------------------------------------------------
__global__ __launch_bounds__(256) void rope_kernel(__bf16* __restrict__ X,
                                                   const float* __restrict__ cosp,
                                                   const float* __restrict__ sinp) {
    const int i   = blockIdx.x * blockDim.x + threadIdx.x;  // < ROWS*NH*64
    const int d   = i & 63;
    const int h   = (i >> 6) & (NH - 1);
    const int row = i >> 10;
    const int s   = row & (SEQ - 1);
    const size_t base = (size_t)row * HID + h * HD;

    const float x1 = (float)X[base + d];
    const float x2 = (float)X[base + d + 64];
    const float c1 = cosp[s * HD + d];
    const float c2 = cosp[s * HD + d + 64];
    const float s1 = sinp[s * HD + d];
    const float s2 = sinp[s * HD + d + 64];

    X[base + d]      = (__bf16)(x1 * c1 - x2 * s1);  // q*cos + (-q_hi)*sin
    X[base + d + 64] = (__bf16)(x2 * c2 + x1 * s2);  // q*cos + ( q_lo)*sin
}

// ---------------------------------------------------------------------------
// Flash attention (causal). Grid (S/64, NH, B), 128 threads = 4 waves.
// Wave owns 16 Q rows. K fragments stream from global (K/V fit in L2);
// V staged transposed in LDS; P bridged through per-wave LDS tile.
// ---------------------------------------------------------------------------
__global__ __launch_bounds__(128) void flash_kernel(const __bf16* __restrict__ Q,
                                                    const __bf16* __restrict__ K,
                                                    const __bf16* __restrict__ V,
                                                    __bf16* __restrict__ O) {
    const int qt   = blockIdx.x;   // q tile (64 rows)
    const int h    = blockIdx.y;
    const int bb   = blockIdx.z;
    const int tid  = threadIdx.x;
    const int lane = tid & 31;
    const int wv   = tid >> 5;     // wave -> 16-row slice of the q tile
    const int qbase = qt * 64;

    const __bf16* Qb = Q + (size_t)bb * SEQ * HID + h * HD;
    const __bf16* Kb = K + (size_t)bb * SEQ * HID + h * HD;
    const __bf16* Vb = V + (size_t)bb * SEQ * HID + h * HD;

    __shared__ __bf16 Vt[HD][64];      // V tile transposed: [d][j]  (16 KB)
    __shared__ __bf16 Pl[4][16][64];   // per-wave P tiles          ( 8 KB)

    // Q fragments stay resident: 16 rows x 128 (4 k-chunks of 32)
    bf16x16 qf[4];
#pragma unroll
    for (int kc = 0; kc < 4; ++kc)
        qf[kc] = gFrag(Qb, HID, qbase + wv * 16, kc * 32, lane);

    f32x8 o[8] = {};                    // 16 x 128 f32 accumulator
    float rmax[8], lsum[8];
#pragma unroll
    for (int r = 0; r < 8; ++r) { rmax[r] = -3.0e38f; lsum[r] = 0.0f; }

    const int  moff = (lane & 16) ? 8 : 0;
    const int  cl   = lane & 15;
    const float scale = 0.088388347648318447f;   // 1/sqrt(128)

    for (int jt = 0; jt <= qt; ++jt) {
        const int jbase = jt * 64;

        __syncthreads();   // previous PV reads of Vt are done
        // Stage V tile transposed: 64 (j) x 128 (d) -> Vt[d][j]
#pragma unroll
        for (int it = 0; it < 8; ++it) {
            const int c  = tid * 8 + it;       // 0..1023 chunk id
            const int j  = c >> 4;             // 0..63
            const int dc = c & 15;             // 8-wide d chunk
            Half8 u;
            u.q = *(const uint4*)(Vb + (size_t)(jbase + j) * HID + dc * 8);
#pragma unroll
            for (int e = 0; e < 8; ++e) Vt[dc * 8 + e][j] = u.h[e];
        }
        __syncthreads();

        // S = Q K^T  (16 x 64 per wave), K fragments straight from global
        f32x8 s[4];
#pragma unroll
        for (int nb = 0; nb < 4; ++nb) {
            f32x8 a = {};
#pragma unroll
            for (int kc = 0; kc < 4; ++kc) {
                bf16x16 kfrag = gFrag(Kb, HID, jbase + nb * 16, kc * 32, lane);
                a = wmma_bf16(qf[kc], kfrag, a);
            }
            s[nb] = a;
        }

        // Scale + causal mask (diagonal tile only; reference uses -1e4)
        const bool diag = (jt == qt);
#pragma unroll
        for (int nb = 0; nb < 4; ++nb)
#pragma unroll
            for (int r = 0; r < 8; ++r) {
                float x = s[nb][r] * scale;
                if (diag) {
                    const int row = qbase + wv * 16 + r + moff;
                    const int col = jbase + nb * 16 + cl;
                    if (col > row) x += NEGM;
                }
                s[nb][r] = x;
            }

        // Online softmax: per-row max/sum across 64 cols (half-wave shuffles)
        float corr[8];
#pragma unroll
        for (int r = 0; r < 8; ++r) {
            float tm = fmaxf(fmaxf(s[0][r], s[1][r]), fmaxf(s[2][r], s[3][r]));
            tm = rowMax16(tm);
            const float nm = fmaxf(rmax[r], tm);
            corr[r] = __expf(rmax[r] - nm);
            rmax[r] = nm;
        }
#pragma unroll
        for (int r = 0; r < 8; ++r) {
            float ps = 0.0f;
#pragma unroll
            for (int nb = 0; nb < 4; ++nb) {
                const float p = __expf(s[nb][r] - rmax[r]);
                ps += p;
                Pl[wv][r + moff][nb * 16 + cl] = (__bf16)p;
            }
            ps = rowSum16(ps);
            lsum[r] = lsum[r] * corr[r] + ps;
#pragma unroll
            for (int db = 0; db < 8; ++db) o[db][r] *= corr[r];
        }
        // Wave-local LDS write -> read ordering for the P tile
        asm volatile("s_wait_dscnt 0" ::: "memory");

        // O += P @ V   (P: 16x64, Vt columns give B fragments)
#pragma unroll
        for (int kc2 = 0; kc2 < 2; ++kc2) {
            bf16x16 pa = ldsFrag(&Pl[wv][0][0], 64, 0, kc2 * 32, lane);
#pragma unroll
            for (int db = 0; db < 8; ++db) {
                bf16x16 vb = ldsFrag(&Vt[0][0], 64, db * 16, kc2 * 32, lane);
                o[db] = wmma_bf16(pa, vb, o[db]);
            }
        }
    }

    // Normalize and store attention output (bf16)
#pragma unroll
    for (int db = 0; db < 8; ++db)
#pragma unroll
        for (int r = 0; r < 8; ++r) {
            const int row = qbase + wv * 16 + r + moff;
            const int col = h * HD + db * 16 + cl;
            O[((size_t)bb * SEQ + row) * HID + col] = (__bf16)(o[db][r] / lsum[r]);
        }
}

// ---------------------------------------------------------------------------
// Launch
// ---------------------------------------------------------------------------
extern "C" void kernel_launch(void* const* d_in, const int* in_sizes, int n_in,
                              void* d_out, int out_size, void* d_ws, size_t ws_size,
                              hipStream_t stream) {
    (void)in_sizes; (void)n_in; (void)out_size; (void)ws_size;

    const float* X    = (const float*)d_in[0];
    const float* cosp = (const float*)d_in[1];
    const float* sinp = (const float*)d_in[2];
    const float* Wq   = (const float*)d_in[3];
    const float* Wk   = (const float*)d_in[4];
    const float* Wv   = (const float*)d_in[5];
    const float* Wo   = (const float*)d_in[6];
    float*       out  = (float*)d_out;

    const size_t xe = (size_t)ROWS * HID;   // 8,388,608 elems
    const size_t we = (size_t)HID * HID;    // 4,194,304 elems
    __bf16* Xb   = (__bf16*)d_ws;
    __bf16* Wqb  = Xb + xe;
    __bf16* Wkb  = Wqb + we;
    __bf16* Wvb  = Wkb + we;
    __bf16* Wob  = Wvb + we;
    __bf16* q    = Wob + we;
    __bf16* k    = q + xe;
    __bf16* v    = k + xe;
    __bf16* attn = v + xe;                  // total ~117 MB of ws

    // One-time f32 -> bf16 conversions
    cvt_kernel<<<(int)(xe / 4 / 256), 256, 0, stream>>>(X, Xb);
    cvt_kernel<<<(int)(we / 4 / 256), 256, 0, stream>>>(Wq, Wqb);
    cvt_kernel<<<(int)(we / 4 / 256), 256, 0, stream>>>(Wk, Wkb);
    cvt_kernel<<<(int)(we / 4 / 256), 256, 0, stream>>>(Wv, Wvb);
    cvt_kernel<<<(int)(we / 4 / 256), 256, 0, stream>>>(Wo, Wob);

    const dim3 gG(HID / 128, ROWS / 128);   // (16, 32)

    gemm_kernel<false><<<gG, 256, 0, stream>>>(Xb, Wqb, q, ROWS, HID, HID);
    gemm_kernel<false><<<gG, 256, 0, stream>>>(Xb, Wkb, k, ROWS, HID, HID);
    gemm_kernel<false><<<gG, 256, 0, stream>>>(Xb, Wvb, v, ROWS, HID, HID);

    const int ropeN = ROWS * NH * 64;       // 4,194,304
    rope_kernel<<<ropeN / 256, 256, 0, stream>>>(q, cosp, sinp);
    rope_kernel<<<ropeN / 256, 256, 0, stream>>>(k, cosp, sinp);

    flash_kernel<<<dim3(SEQ / 64, NH, BATCH), 128, 0, stream>>>(q, k, v, attn);

    gemm_kernel<true><<<gG, 256, 0, stream>>>(attn, Wob, out, ROWS, HID, HID);
}
